// GemmaAttention_41979010351576
// MI455X (gfx1250) — matrix operand order repaired and checked by reading, not compiled, not run on hardware
//
#include <hip/hip_runtime.h>
#include <hip/hip_bf16.h>
#include <type_traits>

typedef __bf16 bf16;
typedef __attribute__((ext_vector_type(16))) __bf16 v16bf;
typedef __attribute__((ext_vector_type(8)))  __bf16 bf16x8;
typedef __attribute__((ext_vector_type(8)))  float  v8f;
typedef int v4i_g __attribute__((vector_size(16)));   // matches builtin param type

#define LOG2E 1.4426950408889634f

// Problem constants
constexpr int CB  = 2;      // batch
constexpr int CS  = 2048;   // seq
constexpr int CD  = 2048;   // model dim
constexpr int CNH = 8;      // heads
constexpr int CHD = 256;    // head dim
constexpr int CM  = CB * CS; // 4096 rows

#if defined(__gfx1250__) && __has_builtin(__builtin_amdgcn_global_load_async_to_lds_b128)
#define USE_ASYNC_LDS 1
#else
#define USE_ASYNC_LDS 0
#endif

// ---------------------------------------------------------------------------
// f32 -> bf16 conversion (grid-stride)
// ---------------------------------------------------------------------------
__global__ void cvt_bf16_kernel(const float* __restrict__ in, bf16* __restrict__ out, int n) {
    for (int i = blockIdx.x * blockDim.x + threadIdx.x; i < n; i += gridDim.x * blockDim.x)
        out[i] = (bf16)in[i];
}

// ---------------------------------------------------------------------------
// BF16 WMMA GEMM: C[M,N] = A[M,K] * B[K,N], row-major, f32 accumulate.
// Block tile: 64(M) x 128(N), 8 waves (256 threads). Each wave: 16(M) x 64(N).
// K loop step 32. A tile staged via async global->LDS DMA; B tile transposed
// through registers into LDS. Next-tile prefetch into L2.
// ---------------------------------------------------------------------------
template <typename OutT>
__global__ void __launch_bounds__(256)
gemm_wmma_bf16(const bf16* __restrict__ A, const bf16* __restrict__ Bm,
               OutT* __restrict__ C, int K, int lda, int ldb, int ldc) {
    __shared__ bf16 ldsA[64 * 32];    // [m][k]
    __shared__ bf16 ldsB[128 * 32];   // transposed: [n][k]

    const int tid  = threadIdx.x;
    const int lane = tid & 31;
    const int wid  = tid >> 5;
    const int wm   = wid & 3;         // wave row   (0..3)
    const int wn   = wid >> 2;        // wave col   (0..1)
    const int m0   = blockIdx.y * 64;
    const int n0   = blockIdx.x * 128;

    const int kg = lane >> 4;         // K-group for A frag / K-half for B frag
    const int ln = lane & 15;

    v8f acc[4] = {};

    const int arow = tid >> 2;        // 0..63
    const int aseg = tid & 3;         // 0..3  (8 elems each)

    for (int k0 = 0; k0 < K; k0 += 32) {
        // L2 prefetch of next k-tile (speculative; safe past end)
        if (k0 + 32 < K) {
            __builtin_prefetch(A + (size_t)(m0 + arow) * lda + k0 + 32 + aseg * 8, 0, 0);
            __builtin_prefetch(Bm + (size_t)(k0 + 32 + (tid >> 4)) * ldb + n0 + (tid & 15) * 8, 0, 0);
        }
        // A tile: 64x32, one 16B transfer per thread
        const bf16* asrc = A + (size_t)(m0 + arow) * lda + k0 + aseg * 8;
#if USE_ASYNC_LDS
        __builtin_amdgcn_global_load_async_to_lds_b128(
            (__attribute__((address_space(1))) v4i_g*)asrc,
            (__attribute__((address_space(3))) v4i_g*)&ldsA[arow * 32 + aseg * 8],
            0, 0);
#else
        *(bf16x8*)&ldsA[arow * 32 + aseg * 8] = *(const bf16x8*)asrc;
#endif
        // B tile: 32x128, store transposed [n][k]
        #pragma unroll
        for (int i = 0; i < 2; ++i) {
            int idx = tid + i * 256;      // 0..511
            int kr  = idx >> 4;           // 0..31
            int seg = idx & 15;           // 8 elems each
            bf16x8 v = *(const bf16x8*)(Bm + (size_t)(k0 + kr) * ldb + n0 + seg * 8);
            #pragma unroll
            for (int j = 0; j < 8; ++j)
                ldsB[(seg * 8 + j) * 32 + kr] = v[j];
        }
#if USE_ASYNC_LDS
#if __has_builtin(__builtin_amdgcn_s_wait_asynccnt)
        __builtin_amdgcn_s_wait_asynccnt(0);
#else
        asm volatile("s_wait_asynccnt 0x0" ::: "memory");
#endif
#endif
        __syncthreads();

        // A fragment: row = wm*16 + ln, elems {kg*8..+7, 16+kg*8..+7}
        v16bf afrag;
        {
            bf16x8 lo = *(const bf16x8*)&ldsA[(wm * 16 + ln) * 32 + kg * 8];
            bf16x8 hi = *(const bf16x8*)&ldsA[(wm * 16 + ln) * 32 + 16 + kg * 8];
            #pragma unroll
            for (int j = 0; j < 8; ++j) { afrag[j] = lo[j]; afrag[8 + j] = hi[j]; }
        }
        #pragma unroll
        for (int sub = 0; sub < 4; ++sub) {
            int n = wn * 64 + sub * 16 + ln;
            v16bf bfrag;
            bf16x8 lo = *(const bf16x8*)&ldsB[n * 32 + kg * 8];
            bf16x8 hi = *(const bf16x8*)&ldsB[n * 32 + 16 + kg * 8];
            #pragma unroll
            for (int j = 0; j < 8; ++j) { bfrag[j] = lo[j]; bfrag[8 + j] = hi[j]; }
            acc[sub] = __builtin_amdgcn_wmma_f32_16x16x32_bf16(
                false, afrag, false, bfrag, (short)0, acc[sub], false, false);
        }
        __syncthreads();
    }

    // Epilogue: C layout — col = ln, rows = r + 8*kg
    #pragma unroll
    for (int sub = 0; sub < 4; ++sub) {
        #pragma unroll
        for (int r = 0; r < 8; ++r) {
            int row = m0 + wm * 16 + r + 8 * kg;
            int col = n0 + wn * 64 + sub * 16 + ln;
            float v = acc[sub][r];
            if constexpr (std::is_same<OutT, float>::value)
                C[(size_t)row * ldc + col] = v;
            else
                C[(size_t)row * ldc + col] = (bf16)v;
        }
    }
}

// ---------------------------------------------------------------------------
// RoPE on Q: read f32 [b*S+s][h*HD+d], write bf16 head-major [b][h][s][d]
// ---------------------------------------------------------------------------
__global__ void rope_q_kernel(const float* __restrict__ Qf, const int* __restrict__ pos,
                              bf16* __restrict__ Qb) {
    int idx = blockIdx.x * blockDim.x + threadIdx.x;   // < B*S*NH*HD = 8388608
    int d = idx & 255;
    int h = (idx >> 8) & 7;
    int s = (idx >> 11) & 2047;
    int b = idx >> 22;
    float p   = (float)pos[b * CS + s];
    float inv = powf(10000.0f, -(float)(d & 127) * (1.0f / 128.0f));
    float ang = p * inv;
    float c = cosf(ang), sn = sinf(ang);
    size_t base = ((size_t)(b * CS + s)) * CD + h * CHD;
    float x   = Qf[base + d];
    float rot = (d < 128) ? -Qf[base + d + 128] : Qf[base + d - 128];
    Qb[((size_t)(b * CNH + h) * CS + s) * CHD + d] = (bf16)(x * c + rot * sn);
}

// ---------------------------------------------------------------------------
// RoPE on K (NKV=1): f32 [b*S+s][d] -> bf16 [b][s][d]
// ---------------------------------------------------------------------------
__global__ void rope_k_kernel(const float* __restrict__ Kf, const int* __restrict__ pos,
                              bf16* __restrict__ Kb) {
    int idx = blockIdx.x * blockDim.x + threadIdx.x;   // < B*S*HD = 1048576
    int d = idx & 255;
    int s = (idx >> 8) & 2047;
    int b = idx >> 19;
    float p   = (float)pos[b * CS + s];
    float inv = powf(10000.0f, -(float)(d & 127) * (1.0f / 128.0f));
    float ang = p * inv;
    float c = cosf(ang), sn = sinf(ang);
    size_t base = ((size_t)(b * CS + s)) * CHD;
    float x   = Kf[base + d];
    float rot = (d < 128) ? -Kf[base + d + 128] : Kf[base + d - 128];
    Kb[base + d] = (bf16)(x * c + rot * sn);
}

// ---------------------------------------------------------------------------
// V transpose: bf16 [b][s][d] -> bf16 [b][d][s]
// ---------------------------------------------------------------------------
__global__ void vtrans_kernel(const bf16* __restrict__ V, bf16* __restrict__ Vt) {
    int idx = blockIdx.x * blockDim.x + threadIdx.x;   // < B*S*HD
    int d = idx & 255;
    int s = (idx >> 8) & 2047;
    int b = idx >> 19;
    Vt[((size_t)(b * CHD + d)) * CS + s] = V[((size_t)(b * CS + s)) * CHD + d];
}

// ---------------------------------------------------------------------------
// Flash attention: 1 wave per (b, h, 16-query tile). Online softmax.
//   Q bf16 [b][h][s][256], K bf16 [b][s][256], Vt bf16 [b][256][s]
//   Out bf16 [b][s][h*256+d]   (row stride 2048, feeds final GEMM)
// K/V fragment loads are software-pipelined one step ahead of the WMMAs.
// ---------------------------------------------------------------------------
__global__ void __launch_bounds__(32)
attn_kernel(const bf16* __restrict__ Qb, const bf16* __restrict__ Kb,
            const bf16* __restrict__ Vt, bf16* __restrict__ Ob) {
    __shared__ bf16 ldsP[16 * 32];

    const int lane = threadIdx.x & 31;
    const int hk = lane >> 4;          // half index (also K-group)
    const int ln = lane & 15;

    int bid = blockIdx.x;
    int qt = bid & 127;                // S/16 = 128 tiles
    int h  = (bid >> 7) & 7;
    int b  = bid >> 10;
    const int q0 = qt * 16;

    // Q fragments: 8 chunks of K-dim 32 (A-frag layout), loaded once
    v16bf qf[8];
    {
        const bf16* qbase = Qb + ((size_t)(b * CNH + h) * CS + q0 + ln) * CHD;
        #pragma unroll
        for (int c = 0; c < 8; ++c) {
            bf16x8 lo = *(const bf16x8*)(qbase + c * 32 + hk * 8);
            bf16x8 hi = *(const bf16x8*)(qbase + c * 32 + 16 + hk * 8);
            #pragma unroll
            for (int j = 0; j < 8; ++j) { qf[c][j] = lo[j]; qf[c][8 + j] = hi[j]; }
        }
    }

    float m_s[8], l_s[8];
    v8f o[16] = {};
    #pragma unroll
    for (int r = 0; r < 8; ++r) { m_s[r] = -3.0e38f; l_s[r] = 0.0f; }

    const float scale = 1.0f / 16.0f;  // 1/sqrt(256)
    const bf16* kbb = Kb + (size_t)b * CS * CHD;
    const bf16* vtb = Vt + (size_t)b * CHD * CS;

    // B-frag loader for QK^T: bf16 k {8*hk..+7, 16+8*hk..+7} of chunk c from one key row
    auto load_kfrag = [&](const bf16* kr, int c) {
        v16bf kf;
        bf16x8 lo = *(const bf16x8*)(kr + c * 32 + hk * 8);
        bf16x8 hi = *(const bf16x8*)(kr + c * 32 + 16 + hk * 8);
        #pragma unroll
        for (int j = 0; j < 8; ++j) { kf[j] = lo[j]; kf[8 + j] = hi[j]; }
        return kf;
    };

    for (int kc = 0; kc < q0 + 16; kc += 32) {
        const bool has2 = (kc + 16) < (q0 + 16);

        // --- scores: two 16-key subtiles, 8 WMMAs each, loads pipelined ---
        v8f s0 = {}, s1 = {};
        {
            const bf16* kr = kbb + (size_t)(kc + ln) * CHD;
            v16bf kf = load_kfrag(kr, 0);
            #pragma unroll
            for (int c = 0; c < 8; ++c) {
                v16bf kn = (c < 7) ? load_kfrag(kr, c + 1) : kf;
                s0 = __builtin_amdgcn_wmma_f32_16x16x32_bf16(
                    false, qf[c], false, kf, (short)0, s0, false, false);
                kf = kn;
            }
        }
        if (has2) {
            const bf16* kr = kbb + (size_t)(kc + 16 + ln) * CHD;
            v16bf kf = load_kfrag(kr, 0);
            #pragma unroll
            for (int c = 0; c < 8; ++c) {
                v16bf kn = (c < 7) ? load_kfrag(kr, c + 1) : kf;
                s1 = __builtin_amdgcn_wmma_f32_16x16x32_bf16(
                    false, qf[c], false, kf, (short)0, s1, false, false);
                kf = kn;
            }
        }

        // --- joint online softmax over the 32-key chunk ---
        float pr0[8], pr1[8];
        #pragma unroll
        for (int r = 0; r < 8; ++r) {
            int q    = q0 + r + 8 * hk;
            int key0 = kc + ln;
            int key1 = kc + 16 + ln;
            float v0 = s0[r] * scale + (key0 <= q ? 0.0f : -1.0e9f);
            float v1 = has2 ? (s1[r] * scale + (key1 <= q ? 0.0f : -1.0e9f)) : -3.0e38f;
            float mx = fmaxf(v0, v1);
            mx = fmaxf(mx, __shfl_xor(mx, 1));
            mx = fmaxf(mx, __shfl_xor(mx, 2));
            mx = fmaxf(mx, __shfl_xor(mx, 4));
            mx = fmaxf(mx, __shfl_xor(mx, 8));
            float mnew  = fmaxf(m_s[r], mx);
            float alpha = exp2f((m_s[r] - mnew) * LOG2E);
            float p0 = exp2f((v0 - mnew) * LOG2E);
            float p1 = has2 ? exp2f((v1 - mnew) * LOG2E) : 0.0f;
            float rs = p0 + p1;
            rs += __shfl_xor(rs, 1);
            rs += __shfl_xor(rs, 2);
            rs += __shfl_xor(rs, 4);
            rs += __shfl_xor(rs, 8);
            l_s[r] = l_s[r] * alpha + rs;
            m_s[r] = mnew;
            pr0[r] = p0; pr1[r] = p1;
            #pragma unroll
            for (int t = 0; t < 16; ++t) o[t][r] *= alpha;
        }

        // --- P (C-layout) -> LDS [q][kk] -> A-frag ---
        #pragma unroll
        for (int r = 0; r < 8; ++r) {
            ldsP[(r + 8 * hk) * 32 + ln]      = (bf16)pr0[r];
            ldsP[(r + 8 * hk) * 32 + 16 + ln] = (bf16)pr1[r];
        }
        __builtin_amdgcn_wave_barrier();   // LDS ops are in-order per wave
        v16bf pf;
        {
            bf16x8 lo = *(const bf16x8*)&ldsP[ln * 32 + hk * 8];
            bf16x8 hi = *(const bf16x8*)&ldsP[ln * 32 + 16 + hk * 8];
            #pragma unroll
            for (int j = 0; j < 8; ++j) { pf[j] = lo[j]; pf[8 + j] = hi[j]; }
        }
        __builtin_amdgcn_wave_barrier();

        // --- O += P * V : 16 WMMAs, V fragment loads pipelined ---
        auto load_vfrag = [&](int t) {
            int d = t * 16 + ln;
            v16bf vf;
            bf16x8 lo = *(const bf16x8*)(vtb + (size_t)d * CS + kc + hk * 8);
            bf16x8 hi = *(const bf16x8*)(vtb + (size_t)d * CS + kc + 16 + hk * 8);
            #pragma unroll
            for (int j = 0; j < 8; ++j) { vf[j] = lo[j]; vf[8 + j] = hi[j]; }
            return vf;
        };
        v16bf vf = load_vfrag(0);
        #pragma unroll
        for (int t = 0; t < 16; ++t) {
            v16bf vn = (t < 15) ? load_vfrag(t + 1) : vf;
            o[t] = __builtin_amdgcn_wmma_f32_16x16x32_bf16(
                false, pf, false, vf, (short)0, o[t], false, false);
            vf = vn;
        }
    }

    // --- finalize: divide by l, write bf16 [b][s][h*256+d] ---
    #pragma unroll
    for (int r = 0; r < 8; ++r) {
        float invl = 1.0f / l_s[r];
        int q = q0 + r + 8 * hk;
        bf16* orow = Ob + ((size_t)(b * CS + q)) * CD + h * CHD;
        #pragma unroll
        for (int t = 0; t < 16; ++t)
            orow[t * 16 + ln] = (bf16)(o[t][r] * invl);
    }
}

// ---------------------------------------------------------------------------
// Launch
// ---------------------------------------------------------------------------
extern "C" void kernel_launch(void* const* d_in, const int* in_sizes, int n_in,
                              void* d_out, int out_size, void* d_ws, size_t ws_size,
                              hipStream_t stream) {
    (void)in_sizes; (void)n_in; (void)out_size; (void)ws_size;
    const float* hidden = (const float*)d_in[0];
    // d_in[1] = attention_mask (deterministic causal; implemented directly)
    const int*   posids = (const int*)d_in[2];
    const float* Wq = (const float*)d_in[3];
    const float* Wk = (const float*)d_in[4];
    const float* Wv = (const float*)d_in[5];
    const float* Wo = (const float*)d_in[6];
    float* out = (float*)d_out;

    char* ws = (char*)d_ws;
    size_t off = 0;
    auto alloc = [&](size_t bytes) {
        void* p = ws + off;
        off += (bytes + 255) & ~(size_t)255;
        return p;
    };

    const size_t nHid = (size_t)CM * CD;        // 8388608
    const size_t nWq  = (size_t)CD * (CNH * CHD);
    const size_t nWkv = (size_t)CD * CHD;       // 524288
    const size_t nKV  = (size_t)CM * CHD;       // 1048576

    bf16* hid_b = (bf16*)alloc(nHid * 2);
    bf16* wq_b  = (bf16*)alloc(nWq  * 2);
    bf16* wk_b  = (bf16*)alloc(nWkv * 2);
    bf16* wv_b  = (bf16*)alloc(nWkv * 2);
    bf16* wo_b  = (bf16*)alloc(nWq  * 2);
    float* q_f  = (float*)alloc(nHid * 4);
    float* k_f  = (float*)alloc(nKV  * 4);
    bf16* v_b   = (bf16*)alloc(nKV * 2);
    bf16* q_b   = (bf16*)alloc(nHid * 2);
    bf16* k_b   = (bf16*)alloc(nKV * 2);
    bf16* vt_b  = (bf16*)alloc(nKV * 2);
    bf16* ao_b  = (bf16*)alloc(nHid * 2);

    // 1) convert to bf16
    cvt_bf16_kernel<<<4096, 256, 0, stream>>>(hidden, hid_b, (int)nHid);
    cvt_bf16_kernel<<<4096, 256, 0, stream>>>(Wq, wq_b, (int)nWq);
    cvt_bf16_kernel<<<1024, 256, 0, stream>>>(Wk, wk_b, (int)nWkv);
    cvt_bf16_kernel<<<1024, 256, 0, stream>>>(Wv, wv_b, (int)nWkv);
    cvt_bf16_kernel<<<4096, 256, 0, stream>>>(Wo, wo_b, (int)nWq);

    // 2) projections (WMMA bf16, f32 accumulate)
    gemm_wmma_bf16<float><<<dim3(CD / 128, CM / 64), 256, 0, stream>>>(
        hid_b, wq_b, q_f, CD, CD, CD, CD);
    gemm_wmma_bf16<float><<<dim3(CHD / 128, CM / 64), 256, 0, stream>>>(
        hid_b, wk_b, k_f, CD, CD, CHD, CHD);
    gemm_wmma_bf16<bf16><<<dim3(CHD / 128, CM / 64), 256, 0, stream>>>(
        hid_b, wv_b, v_b, CD, CD, CHD, CHD);

    // 3) RoPE + relayout
    rope_q_kernel<<<(int)(nHid / 256), 256, 0, stream>>>(q_f, posids, q_b);
    rope_k_kernel<<<(int)(nKV / 256), 256, 0, stream>>>(k_f, posids, k_b);
    vtrans_kernel<<<(int)(nKV / 256), 256, 0, stream>>>(v_b, vt_b);

    // 4) flash attention: B*NH*(S/16) = 2048 waves
    attn_kernel<<<CB * CNH * (CS / 16), 32, 0, stream>>>(q_b, k_b, vt_b, ao_b);

    // 5) output projection
    gemm_wmma_bf16<float><<<dim3(CD / 128, CM / 64), 256, 0, stream>>>(
        ao_b, wo_b, out, CD, CD, CD, CD);
}
